// Model_86483461472452
// MI455X (gfx1250) — matrix operand order repaired and checked by reading, not compiled, not run on hardware
//
#include <hip/hip_runtime.h>

// ---------------------------------------------------------------------------
// Problem constants
// ---------------------------------------------------------------------------
constexpr int kB   = 32;
constexpr int kC0  = 512;
constexpr int kC1  = 1024;
constexpr int kCls = 200;
constexpr int kH   = 28;
constexpr int kW   = 28;
constexpr int kHW  = kH * kW;            // 784
constexpr int kM   = kB * kHW;           // 25088 (GEMM M dimension)

// ---------------------------------------------------------------------------
// Types for CDNA5 WMMA (wave32, V_WMMA_F32_16X16X32_BF16)
// ---------------------------------------------------------------------------
typedef __bf16    v16bf  __attribute__((ext_vector_type(16)));
typedef float     v8f    __attribute__((ext_vector_type(8)));
typedef unsigned  u32x4  __attribute__((ext_vector_type(4)));

union FragBF {                 // 16 bf16 = 8 VGPRs, loaded as two 16B chunks
    u32x4 q[2];
    v16bf v;
};

__device__ __forceinline__ unsigned short f2bf(float f) {
    union { float f; unsigned u; } c; c.f = f;
    unsigned u = c.u;
    u += 0x7fffu + ((u >> 16) & 1u);    // round-to-nearest-even
    return (unsigned short)(u >> 16);
}

// ---------------------------------------------------------------------------
// avg_pool 3x3 stride1 pad1 (sum/9) on fp32 input, emit bf16 features
// ---------------------------------------------------------------------------
__global__ void avgpool3_kernel(const float* __restrict__ x,
                                unsigned short* __restrict__ o, int total) {
    int i = blockIdx.x * blockDim.x + threadIdx.x;
    if (i >= total) return;
    int p = i % kHW;
    int h = p / kW, w = p - h * kW;
    const float* xp = x + (size_t)(i - p);
    float s = 0.f;
#pragma unroll
    for (int dy = -1; dy <= 1; ++dy) {
        int y = h + dy;
        if ((unsigned)y >= (unsigned)kH) continue;
#pragma unroll
        for (int dx = -1; dx <= 1; ++dx) {
            int xx = w + dx;
            if ((unsigned)xx >= (unsigned)kW) continue;
            s += xp[y * kW + xx];
        }
    }
    o[i] = f2bf(s * (1.0f / 9.0f));
}

// ---------------------------------------------------------------------------
// Weight layout conversion: OIHW fp32 -> [O][kh][kw][Ci] bf16  (3x3 convs)
// ---------------------------------------------------------------------------
__global__ void cvt_w3x3_kernel(const float* __restrict__ w,
                                unsigned short* __restrict__ o,
                                int Cin, int total) {
    int i = blockIdx.x * blockDim.x + threadIdx.x;
    if (i >= total) return;                  // i = (oc*9 + khkw)*Cin + ci
    int ci   = i % Cin;
    int t    = i / Cin;
    int khkw = t % 9;
    int oc   = t / 9;
    o[i] = f2bf(w[((size_t)oc * Cin + ci) * 9 + khkw]);
}

// 1x1 conv weights are already [O][Ci]-contiguous: pure dtype convert
__global__ void cvt_w1x1_kernel(const float* __restrict__ w,
                                unsigned short* __restrict__ o, int total) {
    int i = blockIdx.x * blockDim.x + threadIdx.x;
    if (i >= total) return;
    o[i] = f2bf(w[i]);
}

// ---------------------------------------------------------------------------
// Implicit-GEMM conv (pad = KHW/2), bf16 WMMA, f32 accumulate.
//   in : [B][Cin][28][28] bf16   (activations)
//   wgt: [Nout][KHW*KHW*Cin] bf16 (K-order: kh,kw,ci)
// Block = 256 threads (8 wave32), tile M=128 x N=128, K step 32.
// A tile is gathered via unconditional vector-lane loads (clamped base +
// cndmask for padding) and staged through registers -> ds_store_b128.
// B tile is copied global->LDS with the CDNA5 async-copy engine
// (global_load_async_to_lds_b128, ASYNCcnt), no VGPR staging.
// ---------------------------------------------------------------------------
constexpr int TM = 128, TN = 128, TKC = 32;
constexpr int APITCH = 80;   // 64B row + 16B pad -> bank spread for b128 loads
constexpr int BPITCH = 80;

template <int KHW>
__global__ __launch_bounds__(256)
void conv_gemm_kernel(const unsigned short* __restrict__ inp,
                      const unsigned short* __restrict__ wgt,
                      const float* __restrict__ bias,
                      float* __restrict__ outF,            // non-null -> f32 out
                      unsigned short* __restrict__ outH,   // else bf16 out
                      int Cin, int Nout, int relu) {
    __shared__ __align__(16) unsigned char As[2][TM * APITCH];
    __shared__ __align__(16) unsigned char Bs[2][TN * BPITCH];

    const int tid   = threadIdx.x;
    const int tileM = blockIdx.x * TM;
    const int tileN = blockIdx.y * TN;
    const int Ktot  = KHW * KHW * Cin;
    const int nChunks  = Ktot / TKC;
    const int ciChunks = Cin / TKC;

    // --- A-tile loader mapping: thread -> (m row, 16-channel half)
    const int mA    = tileM + (tid & 127);
    const int bImg  = mA / kHW;
    const int hwA   = mA - bImg * kHW;
    const int hA    = hwA / kW;
    const int wA    = hwA - hA * kW;
    const int aHalf = tid >> 7;               // ci sub-offset 16*aHalf

    // --- B-tile loader mapping: 4 threads cover one 64B K-row
    const int bQuad = tid & 3;                // 16B chunk within row
    const int bRow  = tid >> 2;               // rows bRow, bRow+64

    u32x4 aPre[2];

    auto fetchA = [&](int c) {
        // fixed tap (kh,kw) per chunk; 16 channels strided by HW
        const int khkw = c / ciChunks;
        const int ciB  = (c - khkw * ciChunks) * TKC + aHalf * 16;
        int y = hA, x = wA;
        if (KHW == 3) { y = hA + khkw / 3 - 1; x = wA + (khkw % 3) - 1; }
        const bool ok = ((unsigned)y < (unsigned)kH) && ((unsigned)x < (unsigned)kW);
        // Unconditional loads from a per-lane safe address; padded lanes are
        // zero-masked afterwards -> 16 independent global_load_u16, no exec
        // masking, no serialization.
        const unsigned short* pa =
            inp + ((size_t)bImg * Cin + ciB) * kHW + y * kW + x;
        if (!ok) pa = inp;
        unsigned short t[16];
#pragma unroll
        for (int i = 0; i < 16; ++i) t[i] = pa[(size_t)i * kHW];
#pragma unroll
        for (int v = 0; v < 2; ++v) {
            unsigned u[4];
#pragma unroll
            for (int j = 0; j < 4; ++j) {
                const int i = v * 8 + j * 2;
                const unsigned pk = (unsigned)t[i] | ((unsigned)t[i + 1] << 16);
                u[j] = ok ? pk : 0u;
            }
            aPre[v] = (u32x4){u[0], u[1], u[2], u[3]};
        }
    };

    auto stageA = [&](int buf) {
        u32x4* da = (u32x4*)(As[buf] + (tid & 127) * APITCH + aHalf * 32);
        da[0] = aPre[0];
        da[1] = aPre[1];
    };

    // B tile: async global->LDS copy, 2 x 16B per thread (rows bRow, bRow+64).
    // OOB rows (only last N-tile of the 1x1 layer) are skipped; their LDS
    // columns are garbage but those accumulator columns are discarded in the
    // epilogue (GEMM columns are independent).
    auto asyncB = [&](int c, int buf) {
        const int kOff = c * TKC + bQuad * 8;
        const int n0 = tileN + bRow;
        const int n1 = n0 + 64;
        const unsigned lds0 =
            (unsigned)(size_t)(const void*)(Bs[buf] + bRow * BPITCH + bQuad * 16);
        const unsigned lds1 = lds0 + 64 * BPITCH;
        const unsigned short* g0 = wgt + (size_t)n0 * Ktot + kOff;
        const unsigned short* g1 = wgt + (size_t)n1 * Ktot + kOff;
        if (n0 < Nout)
            asm volatile("global_load_async_to_lds_b128 %0, %1, off"
                         :: "v"(lds0), "v"(g0) : "memory");
        if (n1 < Nout)
            asm volatile("global_load_async_to_lds_b128 %0, %1, off"
                         :: "v"(lds1), "v"(g1) : "memory");
    };

    // --- wave tiling: 4 waves along M (32 rows each), 2 along N (64 cols)
    const int lane = tid & 31;
    const int wv   = tid >> 5;
    const int wm   = wv & 3;
    const int wn   = wv >> 2;
    const int lr   = lane & 15;
    const int lh   = lane >> 4;

    v8f acc[2][4] = {};

    auto compute = [&](int buf) {
        FragBF a[2];
#pragma unroll
        for (int i = 0; i < 2; ++i) {
            // A frag (16x32 bf16): lanes 0-15 hold K0-7 & K16-23 of row M=lr,
            // lanes 16-31 hold K8-15 & K24-31.
            const unsigned char* p =
                As[buf] + (wm * 32 + i * 16 + lr) * APITCH + lh * 16;
            a[i].q[0] = *(const u32x4*)p;
            a[i].q[1] = *(const u32x4*)(p + 32);
        }
#pragma unroll
        for (int j = 0; j < 4; ++j) {
            // B frag (32x16 bf16): lane = N, lanes 0-15 hold K0-7 & K16-23,
            // lanes 16-31 hold K8-15 & K24-31.
            FragBF bb;
            const unsigned char* p =
                Bs[buf] + (wn * 64 + j * 16 + lr) * BPITCH + lh * 16;
            bb.q[0] = *(const u32x4*)p;
            bb.q[1] = *(const u32x4*)(p + 32);
#pragma unroll
            for (int i = 0; i < 2; ++i)
                acc[i][j] = __builtin_amdgcn_wmma_f32_16x16x32_bf16(
                    false, a[i].v, false, bb.v, (short)0, acc[i][j],
                    false, false);
        }
    };

    asyncB(0, 0);
    fetchA(0);
    stageA(0);
    asm volatile("s_wait_asynccnt 0x0" ::: "memory");
    __syncthreads();
    for (int c = 0; c < nChunks; ++c) {
        const int  cur  = c & 1;
        const bool more = (c + 1) < nChunks;
        if (more) {
            asyncB(c + 1, cur ^ 1);       // async engine fills next B buffer
            fetchA(c + 1);                // global -> regs for next A buffer
        }
        compute(cur);                     // LDS -> WMMA
        if (more) stageA(cur ^ 1);        // regs -> next A buffer
        asm volatile("s_wait_asynccnt 0x0" ::: "memory");
        __syncthreads();
    }

    // --- epilogue: bias (+ReLU), scatter to NCHW.  D layout: n = lane&15,
    // m = vgpr + 8*(lane>=16).
#pragma unroll
    for (int j = 0; j < 4; ++j) {
        const int nG = tileN + wn * 64 + j * 16 + lr;
        if (nG >= Nout) continue;
        const float bv = bias[nG];
#pragma unroll
        for (int i = 0; i < 2; ++i) {
#pragma unroll
            for (int r = 0; r < 8; ++r) {
                const int mG = tileM + wm * 32 + i * 16 + lh * 8 + r;
                const int bI = mG / kHW;
                const int hw = mG - bI * kHW;
                float v = acc[i][j][r] + bv;
                if (relu) v = fmaxf(v, 0.0f);
                const size_t oi = ((size_t)bI * Nout + nG) * kHW + hw;
                if (outF) outF[oi] = v;
                else      outH[oi] = f2bf(v);
            }
        }
    }
}

// ---------------------------------------------------------------------------
// Spatial mean over HxW: one block per (b, class)
// ---------------------------------------------------------------------------
__global__ __launch_bounds__(256)
void mean_hw_kernel(const float* __restrict__ src, float* __restrict__ dst) {
    const int bk = blockIdx.x;
    const float* p = src + (size_t)bk * kHW;
    float s = 0.f;
    for (int i = threadIdx.x; i < kHW; i += 256) s += p[i];
#pragma unroll
    for (int off = 16; off > 0; off >>= 1) s += __shfl_xor(s, off, 32);
    __shared__ float red[8];
    if ((threadIdx.x & 31) == 0) red[threadIdx.x >> 5] = s;
    __syncthreads();
    if (threadIdx.x == 0) {
        float t = 0.f;
#pragma unroll
        for (int i = 0; i < 8; ++i) t += red[i];
        dst[bk] = t * (1.0f / kHW);
    }
}

// ---------------------------------------------------------------------------
// Attention map -> threshold mask. One block per sample.
// atten = max(0, max over labeled channels), min-max normalized over HW.
// ---------------------------------------------------------------------------
__global__ __launch_bounds__(256)
void atten_mask_kernel(const float* __restrict__ outA,
                       const int* __restrict__ label,
                       float* __restrict__ mask) {
    const int b = blockIdx.x;
    __shared__ float att[kHW];
    __shared__ unsigned char flag[kCls];
    __shared__ float redn[8], redx[8];
    __shared__ float smn, smx;

    for (int c = threadIdx.x; c < kCls; c += 256)
        flag[c] = (label[b * kCls + c] > 0) ? 1 : 0;
    __syncthreads();

    float lmin = 3.4e38f, lmax = -3.4e38f;
    for (int p = threadIdx.x; p < kHW; p += 256) {
        float a = 0.f;   // seed 0 == maximum(sel.max, 0)
        const float* base = outA + (size_t)b * kCls * kHW + p;
        for (int c = 0; c < kCls; ++c)
            if (flag[c]) a = fmaxf(a, base[(size_t)c * kHW]);
        att[p] = a;
        lmin = fminf(lmin, a);
        lmax = fmaxf(lmax, a);
    }
#pragma unroll
    for (int off = 16; off > 0; off >>= 1) {
        lmin = fminf(lmin, __shfl_xor(lmin, off, 32));
        lmax = fmaxf(lmax, __shfl_xor(lmax, off, 32));
    }
    if ((threadIdx.x & 31) == 0) {
        redn[threadIdx.x >> 5] = lmin;
        redx[threadIdx.x >> 5] = lmax;
    }
    __syncthreads();
    if (threadIdx.x == 0) {
        float mn = redn[0], mx = redx[0];
        for (int i = 1; i < 8; ++i) {
            mn = fminf(mn, redn[i]);
            mx = fmaxf(mx, redx[i]);
        }
        smn = mn; smx = mx;
    }
    __syncthreads();
    const float inv = 1.0f / (smx - smn);
    for (int p = threadIdx.x; p < kHW; p += 256) {
        float an = (att[p] - smn) * inv;
        mask[b * kHW + p] = (an >= 0.6f) ? 0.f : 1.f;
    }
}

// feats_erased = feats * mask (mask is 0/1 -> exact select on bf16)
__global__ void erase_kernel(const unsigned short* __restrict__ feats,
                             const float* __restrict__ mask,
                             unsigned short* __restrict__ o, int total) {
    int i = blockIdx.x * blockDim.x + threadIdx.x;
    if (i >= total) return;
    int p = i % kHW;
    int b = i / (kC0 * kHW);
    o[i] = (mask[b * kHW + p] != 0.f) ? feats[i] : (unsigned short)0;
}

// ---------------------------------------------------------------------------
// Host orchestration (graph-capture safe: only kernel launches on `stream`)
// ---------------------------------------------------------------------------
extern "C" void kernel_launch(void* const* d_in, const int* in_sizes, int n_in,
                              void* d_out, int out_size, void* d_ws,
                              size_t ws_size, hipStream_t stream) {
    const float* x    = (const float*)d_in[0];
    const int*  label = (const int*)d_in[1];
    const float* w1a = (const float*)d_in[2];
    const float* b1a = (const float*)d_in[3];
    const float* w2a = (const float*)d_in[4];
    const float* b2a = (const float*)d_in[5];
    const float* w3a = (const float*)d_in[6];
    const float* b3a = (const float*)d_in[7];
    const float* w1b = (const float*)d_in[8];
    const float* b1b = (const float*)d_in[9];
    const float* w2b = (const float*)d_in[10];
    const float* b2b = (const float*)d_in[11];
    const float* w3b = (const float*)d_in[12];
    const float* b3b = (const float*)d_in[13];
    float* out = (float*)d_out;

    unsigned char* ws = (unsigned char*)d_ws;
    size_t off = 0;
    auto alloc = [&](size_t bytes) -> void* {
        void* p = ws + off;
        off += (bytes + 255) & ~((size_t)255);
        return p;
    };

    const size_t featElems = (size_t)kB * kC0 * kHW;     // 12.8M
    const size_t actElems  = (size_t)kM * kC1;           // 25.7M
    unsigned short* featsbf = (unsigned short*)alloc(featElems * 2);
    unsigned short* featsE  = (unsigned short*)alloc(featElems * 2);
    unsigned short* w1abf = (unsigned short*)alloc((size_t)kC1 * 9 * kC0 * 2);
    unsigned short* w2abf = (unsigned short*)alloc((size_t)kC1 * 9 * kC1 * 2);
    unsigned short* w3abf = (unsigned short*)alloc((size_t)kCls * kC1 * 2);
    unsigned short* w1bbf = (unsigned short*)alloc((size_t)kC1 * 9 * kC0 * 2);
    unsigned short* w2bbf = (unsigned short*)alloc((size_t)kC1 * 9 * kC1 * 2);
    unsigned short* w3bbf = (unsigned short*)alloc((size_t)kCls * kC1 * 2);
    unsigned short* act1  = (unsigned short*)alloc(actElems * 2);
    unsigned short* act2  = (unsigned short*)alloc(actElems * 2);
    float* outC  = (float*)alloc((size_t)kB * kCls * kHW * 4);
    float* maskb = (float*)alloc((size_t)kB * kHW * 4);

    const int totF = kB * kC0 * kHW;
    avgpool3_kernel<<<(totF + 255) / 256, 256, 0, stream>>>(x, featsbf, totF);

    const int t1 = kC1 * 9 * kC0;
    const int t2 = kC1 * 9 * kC1;
    const int t3 = kCls * kC1;
    cvt_w3x3_kernel<<<(t1 + 255) / 256, 256, 0, stream>>>(w1a, w1abf, kC0, t1);
    cvt_w3x3_kernel<<<(t2 + 255) / 256, 256, 0, stream>>>(w2a, w2abf, kC1, t2);
    cvt_w1x1_kernel<<<(t3 + 255) / 256, 256, 0, stream>>>(w3a, w3abf, t3);
    cvt_w3x3_kernel<<<(t1 + 255) / 256, 256, 0, stream>>>(w1b, w1bbf, kC0, t1);
    cvt_w3x3_kernel<<<(t2 + 255) / 256, 256, 0, stream>>>(w2b, w2bbf, kC1, t2);
    cvt_w1x1_kernel<<<(t3 + 255) / 256, 256, 0, stream>>>(w3b, w3bbf, t3);

    dim3 grid12(kM / TM, kC1 / TN);                     // 196 x 8
    dim3 grid3(kM / TM, (kCls + TN - 1) / TN);          // 196 x 2

    // --- classifier A
    conv_gemm_kernel<3><<<grid12, 256, 0, stream>>>(
        featsbf, w1abf, b1a, nullptr, act1, kC0, kC1, 1);
    conv_gemm_kernel<3><<<grid12, 256, 0, stream>>>(
        act1, w2abf, b2a, nullptr, act2, kC1, kC1, 1);
    conv_gemm_kernel<1><<<grid3, 256, 0, stream>>>(
        act2, w3abf, b3a, outC, nullptr, kC1, kCls, 0);
    mean_hw_kernel<<<kB * kCls, 256, 0, stream>>>(outC, out);

    // --- attention erase
    atten_mask_kernel<<<kB, 256, 0, stream>>>(outC, label, maskb);
    erase_kernel<<<(totF + 255) / 256, 256, 0, stream>>>(featsbf, maskb,
                                                         featsE, totF);

    // --- classifier B
    conv_gemm_kernel<3><<<grid12, 256, 0, stream>>>(
        featsE, w1bbf, b1b, nullptr, act1, kC0, kC1, 1);
    conv_gemm_kernel<3><<<grid12, 256, 0, stream>>>(
        act1, w2bbf, b2b, nullptr, act2, kC1, kC1, 1);
    conv_gemm_kernel<1><<<grid3, 256, 0, stream>>>(
        act2, w3bbf, b3b, outC, nullptr, kC1, kCls, 0);
    mean_hw_kernel<<<kB * kCls, 256, 0, stream>>>(outC, out + kB * kCls);
}
